// MPSGNN_12945031430849
// MI455X (gfx1250) — compile-verified, matrix operand order checked by reading, not run on hardware
//
#include <hip/hip_runtime.h>

#define DEV __device__ __forceinline__

typedef __attribute__((ext_vector_type(16))) __bf16 v16bf;
typedef __attribute__((ext_vector_type(8)))  float  v8f;

constexpr int Nn  = 30000;
constexpr int INc = 128;
constexpr int Hc  = 64;
constexpr int OUTc= 64;
constexpr int Mc  = 4;
constexpr int Ec  = 480000;
constexpr int FFc = 2048;

DEV unsigned short f2bf(float f) {
    unsigned int u = __float_as_uint(f);
    unsigned int r = u + 0x7FFFu + ((u >> 16) & 1u);   // round-to-nearest-even
    return (unsigned short)(r >> 16);
}

union BFrag { v16bf v; unsigned short u[16]; };

DEV v8f zero8() { v8f z;
#pragma unroll
    for (int i = 0; i < 8; ++i) z[i] = 0.f;
    return z; }

// A-fragment (16x32 bf16) built on the fly from row-major f32 (rows = M).
// ISA 7.12.2 layout: lanes 0-15 row=lane, halves 0-7 = K kb..kb+7, halves 8-15 = K kb+16..kb+23,
// kb = k0 (+8 for lanes 16-31).
DEV v16bf fragA_f32(const float* __restrict__ base, int ld, int k0, int lane) {
    int row = lane & 15;
    int kb  = k0 + ((lane >> 4) << 3);
    const float* p = base + (size_t)row * ld + kb;
    BFrag f;
#pragma unroll
    for (int i = 0; i < 8; ++i) { f.u[i] = f2bf(p[i]); f.u[8 + i] = f2bf(p[16 + i]); }
    return f.v;
}

// Fragment from row-major bf16 (ushort). For B of X@W^T pass row = nbase + (lane&15)
// (B mirrors A with N on lanes); for an LDS-staged A pass row = lane&15.
DEV v16bf frag_u16(const unsigned short* __restrict__ W, int ld, int row, int k0, int lane) {
    int kb = k0 + ((lane >> 4) << 3);
    const unsigned short* p = W + (size_t)row * ld + kb;
    BFrag f;
#pragma unroll
    for (int i = 0; i < 8; ++i) { f.u[i] = p[i]; f.u[8 + i] = p[16 + i]; }
    return f.v;
}

DEV v8f wmma_bf16(v16bf a, v16bf b, v8f c) {
    return __builtin_amdgcn_wmma_f32_16x16x32_bf16(false, a, false, b, (short)0, c, false, false);
}

// ---------------- prep kernels ----------------
__global__ void cvt_bf16_kernel(const float* __restrict__ s, unsigned short* __restrict__ d, int n) {
    int i = blockIdx.x * blockDim.x + threadIdx.x;
    if (i < n) d[i] = f2bf(s[i]);
}
__global__ void addcvt_kernel(const float* __restrict__ a, const float* __restrict__ b,
                              unsigned short* __restrict__ d, int n) {
    int i = blockIdx.x * blockDim.x + threadIdx.x;
    if (i < n) d[i] = f2bf(a[i] + b[i]);
}
__global__ void bias3_kernel(const float* __restrict__ a, const float* __restrict__ b,
                             const float* __restrict__ c, float* __restrict__ d, int n) {
    int i = blockIdx.x * blockDim.x + threadIdx.x;
    if (i < n) d[i] = a[i] + b[i] + c[i];
}

// ---------------- scatter-add: agg[m, src] += x[dst] ----------------
__global__ void scatter_kernel(const float* __restrict__ x, const int* __restrict__ ei,
                               float* __restrict__ agg) {
    int t = blockIdx.x * blockDim.x + threadIdx.x;   // < M*E*32
    int m = t / (Ec * 32);
    int r = t - m * (Ec * 32);
    int e = r >> 5;
    int c = r & 31;                                   // float4 chunk within 128-ch row
    int src = ei[(size_t)(m * 2) * Ec + e];
    int dst = ei[(size_t)(m * 2 + 1) * Ec + e];
    float4 v = ((const float4*)x)[(size_t)dst * (INc / 4) + c];
    float* p = agg + (size_t)m * Nn * INc + (size_t)src * INc + c * 4;
    atomicAdd(p + 0, v.x); atomicAdd(p + 1, v.y);
    atomicAdd(p + 2, v.z); atomicAdd(p + 3, v.w);
}

// ---------------- fused GNN ----------------
__global__ __launch_bounds__(128) void gnn_kernel(
    const float* __restrict__ x, const float* __restrict__ agg,
    const unsigned short* __restrict__ Wl, const unsigned short* __restrict__ W01,
    const unsigned short* __restrict__ Wo, const float* __restrict__ bsum,
    const float* __restrict__ bo, const float* __restrict__ mpw,
    float* __restrict__ e) {
    __shared__ unsigned short sh[4][16 * 64];
    int lane = threadIdx.x & 31, wv = threadIdx.x >> 5;
    int w = blockIdx.x * 4 + wv;
    int m = w / (Nn / 16);
    int tile = w - m * (Nn / 16);
    int rowbase = tile * 16;
    const float* aggm = agg + (size_t)m * Nn * INc + (size_t)rowbase * INc;
    const float* xm   = x   + (size_t)rowbase * INc;
    const unsigned short* wl  = Wl  + (size_t)m * Hc * INc;
    const unsigned short* w01 = W01 + (size_t)m * Hc * INc;
    const unsigned short* wo  = Wo  + (size_t)m * OUTc * Hc;

    int colb = lane & 15, rowb = (lane >> 4) * 8;
    v8f acc1[4];
#pragma unroll
    for (int i = 0; i < 4; ++i) acc1[i] = zero8();
#pragma unroll
    for (int k0 = 0; k0 < INc; k0 += 32) {
        v16bf a = fragA_f32(aggm, INc, k0, lane);
#pragma unroll
        for (int nt = 0; nt < 4; ++nt)
            acc1[nt] = wmma_bf16(a, frag_u16(wl, INc, nt * 16 + colb, k0, lane), acc1[nt]);
        v16bf a2 = fragA_f32(xm, INc, k0, lane);
#pragma unroll
        for (int nt = 0; nt < 4; ++nt)
            acc1[nt] = wmma_bf16(a2, frag_u16(w01, INc, nt * 16 + colb, k0, lane), acc1[nt]);
    }
#pragma unroll
    for (int nt = 0; nt < 4; ++nt) {
        int col = nt * 16 + colb;
        float bb = bsum[m * Hc + col];
#pragma unroll
        for (int r = 0; r < 8; ++r) {
            float v = acc1[nt][r] + bb;
            v = v > 0.f ? v : 0.f;
            sh[wv][(rowb + r) * 64 + col] = f2bf(v);
        }
    }
    __builtin_amdgcn_wave_barrier();
    v8f acc2[4];
#pragma unroll
    for (int i = 0; i < 4; ++i) acc2[i] = zero8();
#pragma unroll
    for (int k0 = 0; k0 < Hc; k0 += 32) {
        v16bf a = frag_u16(sh[wv], 64, colb, k0, lane);
#pragma unroll
        for (int nt = 0; nt < 4; ++nt)
            acc2[nt] = wmma_bf16(a, frag_u16(wo, Hc, nt * 16 + colb, k0, lane), acc2[nt]);
    }
    float wgt = mpw[m];
#pragma unroll
    for (int nt = 0; nt < 4; ++nt) {
        int col = nt * 16 + colb;
        float bb = bo[m * OUTc + col];
#pragma unroll
        for (int r = 0; r < 8; ++r) {
            int node = rowbase + rowb + r;
            e[((size_t)node * Mc + m) * OUTc + col] = (acc2[nt][r] + bb) * wgt;
        }
    }
}

// ---------------- QKV projection ----------------
__global__ __launch_bounds__(128) void qkv_kernel(const float* __restrict__ e,
    const unsigned short* __restrict__ tinw, const float* __restrict__ tinb,
    float* __restrict__ qkv, int l) {
    int lane = threadIdx.x & 31, wv = threadIdx.x >> 5;
    int w = blockIdx.x * 4 + wv;
    int rowbase = w * 16;
    const unsigned short* W = tinw + (size_t)l * 192 * 64;
    const float* bi = tinb + l * 192;
    int colb = lane & 15, rowb = (lane >> 4) * 8;
    v8f acc[12];
#pragma unroll
    for (int i = 0; i < 12; ++i) acc[i] = zero8();
#pragma unroll
    for (int k0 = 0; k0 < 64; k0 += 32) {
        v16bf a = fragA_f32(e + (size_t)rowbase * 64, 64, k0, lane);
#pragma unroll
        for (int nt = 0; nt < 12; ++nt)
            acc[nt] = wmma_bf16(a, frag_u16(W, 64, nt * 16 + colb, k0, lane), acc[nt]);
    }
#pragma unroll
    for (int nt = 0; nt < 12; ++nt) {
        int col = nt * 16 + colb;
        float bb = bi[col];
#pragma unroll
        for (int r = 0; r < 8; ++r)
            qkv[(size_t)(rowbase + rowb + r) * 192 + col] = acc[nt][r] + bb;
    }
}

// ---------------- attention core ----------------
__global__ __launch_bounds__(128) void attn_kernel(const float* __restrict__ qkv,
                                                   float* __restrict__ o) {
    int lane = threadIdx.x & 31, wv = threadIdx.x >> 5;
    int node = blockIdx.x * 4 + wv;
    int h = lane >> 2, qi = lane & 3;
    const float* base = qkv + (size_t)node * 4 * 192;
    float q[8];
#pragma unroll
    for (int d = 0; d < 8; ++d) q[d] = base[qi * 192 + h * 8 + d];
    float s[4];
#pragma unroll
    for (int j = 0; j < 4; ++j) {
        float t = 0.f;
#pragma unroll
        for (int d = 0; d < 8; ++d) t += q[d] * base[j * 192 + 64 + h * 8 + d];
        s[j] = t * 0.3535533905932738f;
    }
    float mx = fmaxf(fmaxf(s[0], s[1]), fmaxf(s[2], s[3]));
    float sum = 0.f;
#pragma unroll
    for (int j = 0; j < 4; ++j) { s[j] = __expf(s[j] - mx); sum += s[j]; }
    float inv = 1.f / sum;
#pragma unroll
    for (int d = 0; d < 8; ++d) {
        float t = 0.f;
#pragma unroll
        for (int j = 0; j < 4; ++j) t += s[j] * base[j * 192 + 128 + h * 8 + d];
        o[(size_t)(node * 4 + qi) * 64 + h * 8 + d] = t * inv;
    }
}

// ---------------- out-proj + residual + LayerNorm ----------------
__global__ __launch_bounds__(128) void oproj_ln_kernel(const float* __restrict__ o,
    const unsigned short* __restrict__ tow, const float* __restrict__ tob,
    const float* __restrict__ g, const float* __restrict__ bln,
    float* __restrict__ e, int l) {
    __shared__ float shf[4][16 * 64];
    int lane = threadIdx.x & 31, wv = threadIdx.x >> 5;
    int w = blockIdx.x * 4 + wv;
    int rowbase = w * 16;
    const unsigned short* W = tow + (size_t)l * 64 * 64;
    int colb = lane & 15, rowb = (lane >> 4) * 8;
    v8f acc[4];
#pragma unroll
    for (int i = 0; i < 4; ++i) acc[i] = zero8();
#pragma unroll
    for (int k0 = 0; k0 < 64; k0 += 32) {
        v16bf a = fragA_f32(o + (size_t)rowbase * 64, 64, k0, lane);
#pragma unroll
        for (int nt = 0; nt < 4; ++nt)
            acc[nt] = wmma_bf16(a, frag_u16(W, 64, nt * 16 + colb, k0, lane), acc[nt]);
    }
#pragma unroll
    for (int nt = 0; nt < 4; ++nt)
#pragma unroll
        for (int r = 0; r < 8; ++r)
            shf[wv][(rowb + r) * 64 + nt * 16 + colb] = acc[nt][r];
    __builtin_amdgcn_wave_barrier();
    int row = lane & 15, c0 = (lane >> 4) * 32;
    const float* bi = tob + l * 64;
    size_t erow = (size_t)(rowbase + row) * 64;
    float s1 = 0.f, s2 = 0.f;
#pragma unroll
    for (int j = 0; j < 32; ++j) {
        float t = shf[wv][row * 64 + c0 + j] + bi[c0 + j] + e[erow + c0 + j];
        s1 += t; s2 += t * t;
    }
    s1 += __shfl_xor(s1, 16, 32);
    s2 += __shfl_xor(s2, 16, 32);
    float mean = s1 * (1.f / 64.f);
    float var  = s2 * (1.f / 64.f) - mean * mean;
    float rs   = rsqrtf(var + 1e-5f);
#pragma unroll
    for (int j = 0; j < 32; ++j) {
        int c = c0 + j;
        float t = shf[wv][row * 64 + c] + bi[c] + e[erow + c];
        e[erow + c] = (t - mean) * rs * g[l * 64 + c] + bln[l * 64 + c];
    }
}

// ---------------- fused FFN: 32-row wave tiles, FF dim streamed in chunks of 32 ----------------
// Two 16-row sub-tiles share every weight fragment -> 32 FLOP/byte from L2 (was 16).
__global__ __launch_bounds__(128) void ff_kernel(
    const unsigned short* __restrict__ ff1w, const float* __restrict__ ff1b,
    const unsigned short* __restrict__ ff2w, const float* __restrict__ ff2b,
    const float* __restrict__ g, const float* __restrict__ bln,
    float* __restrict__ e, int l) {
    __shared__ unsigned short shh[4][32 * 32];   // H staging, bf16, per-wave 2KB
    __shared__ float shf[4][32 * 64];            // output staging for LN, per-wave 8KB
    int lane = threadIdx.x & 31, wv = threadIdx.x >> 5;
    int w = blockIdx.x * 4 + wv;
    int rowbase = w * 32;
    if (rowbase >= Nn * Mc) return;              // only per-wave barriers below: safe
    const unsigned short* W1 = ff1w + (size_t)l * FFc * 64;
    const unsigned short* W2 = ff2w + (size_t)l * 64 * FFc;
    const float* b1 = ff1b + (size_t)l * FFc;
    int colb = lane & 15, rowb = (lane >> 4) * 8;

    // loop-invariant A fragments for both 16-row sub-tiles
    v16bf ea[2][2];
#pragma unroll
    for (int st = 0; st < 2; ++st) {
        const float* eb = e + (size_t)(rowbase + st * 16) * 64;
        ea[st][0] = fragA_f32(eb, 64, 0,  lane);
        ea[st][1] = fragA_f32(eb, 64, 32, lane);
    }

    v8f acc[2][4];
#pragma unroll
    for (int st = 0; st < 2; ++st)
#pragma unroll
        for (int i = 0; i < 4; ++i) acc[st][i] = zero8();

    for (int c0 = 0; c0 < FFc; c0 += 32) {
        v8f h[2][2];
#pragma unroll
        for (int st = 0; st < 2; ++st) { h[st][0] = zero8(); h[st][1] = zero8(); }
        // GEMM1: each W1 fragment feeds both sub-tiles
#pragma unroll
        for (int half = 0; half < 2; ++half) {
            v16bf bA = frag_u16(W1, 64, c0 + half * 16 + colb, 0,  lane);
            v16bf bB = frag_u16(W1, 64, c0 + half * 16 + colb, 32, lane);
#pragma unroll
            for (int st = 0; st < 2; ++st) {
                h[st][half] = wmma_bf16(ea[st][0], bA, h[st][half]);
                h[st][half] = wmma_bf16(ea[st][1], bB, h[st][half]);
            }
        }
        // bias + relu, re-stage as bf16 A-fragment source (32 rows x 32 cols)
#pragma unroll
        for (int st = 0; st < 2; ++st)
#pragma unroll
            for (int half = 0; half < 2; ++half) {
                v8f hh = h[st][half];
                int cl = half * 16 + colb;
                float bb = b1[c0 + cl];
#pragma unroll
                for (int r = 0; r < 8; ++r) {
                    float v = hh[r] + bb;
                    v = v > 0.f ? v : 0.f;
                    shh[wv][(st * 16 + rowb + r) * 32 + cl] = f2bf(v);
                }
            }
        __builtin_amdgcn_wave_barrier();
        v16bf a2[2];
#pragma unroll
        for (int st = 0; st < 2; ++st)
            a2[st] = frag_u16(shh[wv] + st * 16 * 32, 32, colb, 0, lane);
        // GEMM2: each W2 fragment feeds both sub-tiles
#pragma unroll
        for (int nt = 0; nt < 4; ++nt) {
            v16bf b2 = frag_u16(W2, FFc, nt * 16 + colb, c0, lane);
#pragma unroll
            for (int st = 0; st < 2; ++st)
                acc[st][nt] = wmma_bf16(a2[st], b2, acc[st][nt]);
        }
        __builtin_amdgcn_wave_barrier();
    }
    // stage outputs, then residual + LN (ln2) over all 32 rows
#pragma unroll
    for (int st = 0; st < 2; ++st)
#pragma unroll
        for (int nt = 0; nt < 4; ++nt)
#pragma unroll
            for (int r = 0; r < 8; ++r)
                shf[wv][(st * 16 + rowb + r) * 64 + nt * 16 + colb] = acc[st][nt][r];
    __builtin_amdgcn_wave_barrier();
    const float* bi = ff2b + l * 64;
    int c0b = (lane >> 4) * 32;
#pragma unroll
    for (int st = 0; st < 2; ++st) {
        int row = st * 16 + (lane & 15);
        size_t erow = (size_t)(rowbase + row) * 64;
        float s1 = 0.f, s2 = 0.f;
#pragma unroll
        for (int j = 0; j < 32; ++j) {
            float t = shf[wv][row * 64 + c0b + j] + bi[c0b + j] + e[erow + c0b + j];
            s1 += t; s2 += t * t;
        }
        s1 += __shfl_xor(s1, 16, 32);
        s2 += __shfl_xor(s2, 16, 32);
        float mean = s1 * (1.f / 64.f);
        float var  = s2 * (1.f / 64.f) - mean * mean;
        float rs   = rsqrtf(var + 1e-5f);
#pragma unroll
        for (int j = 0; j < 32; ++j) {
            int c = c0b + j;
            float t = shf[wv][row * 64 + c] + bi[c] + e[erow + c];
            e[erow + c] = (t - mean) * rs * g[l * 64 + c] + bln[l * 64 + c];
        }
    }
}

// ---------------- mean-pool + regressor head ----------------
__global__ void head_kernel(const float* __restrict__ e, const float* __restrict__ r1w,
                            const float* __restrict__ r1b, const float* __restrict__ r2w,
                            const float* __restrict__ r2b, float* __restrict__ out) {
    int n = blockIdx.x * blockDim.x + threadIdx.x;
    if (n >= Nn) return;
    const float* er = e + (size_t)n * Mc * OUTc;
    float p[64];
#pragma unroll
    for (int c = 0; c < 64; ++c)
        p[c] = (er[c] + er[64 + c] + er[128 + c] + er[192 + c]) * 0.25f;
    float o = r2b[0];
    for (int j = 0; j < 64; ++j) {
        float s = r1b[j];
        const float* wr = r1w + j * 64;
#pragma unroll
        for (int c = 0; c < 64; ++c) s += p[c] * wr[c];
        s = s > 0.f ? s : 0.f;
        o += s * r2w[j];
    }
    out[n] = o;
}

extern "C" void kernel_launch(void* const* d_in, const int* in_sizes, int n_in,
                              void* d_out, int out_size, void* d_ws, size_t ws_size,
                              hipStream_t stream) {
    const float* x    = (const float*)d_in[0];
    const int*   ei   = (const int*)  d_in[1];
    const float* mpw  = (const float*)d_in[2];
    const float* W0   = (const float*)d_in[3];
    const float* b0   = (const float*)d_in[4];
    const float* Wl   = (const float*)d_in[5];
    const float* bl   = (const float*)d_in[6];
    const float* W1   = (const float*)d_in[7];
    const float* b1   = (const float*)d_in[8];
    const float* Wo   = (const float*)d_in[9];
    const float* bo   = (const float*)d_in[10];
    const float* tinw = (const float*)d_in[11];
    const float* tinb = (const float*)d_in[12];
    const float* tow  = (const float*)d_in[13];
    const float* tob  = (const float*)d_in[14];
    const float* ln1g = (const float*)d_in[15];
    const float* ln1b = (const float*)d_in[16];
    const float* ff1w = (const float*)d_in[17];
    const float* ff1b = (const float*)d_in[18];
    const float* ff2w = (const float*)d_in[19];
    const float* ff2b = (const float*)d_in[20];
    const float* ln2g = (const float*)d_in[21];
    const float* ln2b = (const float*)d_in[22];
    const float* r1w  = (const float*)d_in[23];
    const float* r1b  = (const float*)d_in[24];
    const float* r2w  = (const float*)d_in[25];
    const float* r2b  = (const float*)d_in[26];
    float* out = (float*)d_out;

    char* ws = (char*)d_ws;
    size_t off = 0;
    auto alloc = [&](size_t b) { size_t o = off; off += (b + 255) & ~(size_t)255; return o; };
    size_t o_regA = alloc((size_t)Nn * Mc * 192 * 4);   // agg aliased by qkv
    size_t o_e    = alloc((size_t)Nn * Mc * 64 * 4);
    size_t o_o    = alloc((size_t)Nn * Mc * 64 * 4);
    size_t o_wl   = alloc((size_t)Mc * Hc * INc * 2);
    size_t o_w01  = alloc((size_t)Mc * Hc * INc * 2);
    size_t o_wo   = alloc((size_t)Mc * OUTc * Hc * 2);
    size_t o_tinw = alloc((size_t)2 * 192 * 64 * 2);
    size_t o_tow  = alloc((size_t)2 * 64 * 64 * 2);
    size_t o_ff1  = alloc((size_t)2 * FFc * 64 * 2);
    size_t o_ff2  = alloc((size_t)2 * 64 * FFc * 2);
    size_t o_bsum = alloc((size_t)Mc * Hc * 4);

    float* agg  = (float*)(ws + o_regA);
    float* qkvb = (float*)(ws + o_regA);
    float* ebuf = (float*)(ws + o_e);
    float* obuf = (float*)(ws + o_o);
    unsigned short* wl_b   = (unsigned short*)(ws + o_wl);
    unsigned short* w01_b  = (unsigned short*)(ws + o_w01);
    unsigned short* wo_b   = (unsigned short*)(ws + o_wo);
    unsigned short* tinw_b = (unsigned short*)(ws + o_tinw);
    unsigned short* tow_b  = (unsigned short*)(ws + o_tow);
    unsigned short* ff1_b  = (unsigned short*)(ws + o_ff1);
    unsigned short* ff2_b  = (unsigned short*)(ws + o_ff2);
    float* bsum = (float*)(ws + o_bsum);

    cvt_bf16_kernel<<<(Mc * Hc * INc + 255) / 256, 256, 0, stream>>>(Wl, wl_b, Mc * Hc * INc);
    cvt_bf16_kernel<<<(Mc * OUTc * Hc + 255) / 256, 256, 0, stream>>>(Wo, wo_b, Mc * OUTc * Hc);
    cvt_bf16_kernel<<<(2 * 192 * 64 + 255) / 256, 256, 0, stream>>>(tinw, tinw_b, 2 * 192 * 64);
    cvt_bf16_kernel<<<(2 * 64 * 64 + 255) / 256, 256, 0, stream>>>(tow, tow_b, 2 * 64 * 64);
    cvt_bf16_kernel<<<(2 * FFc * 64 + 255) / 256, 256, 0, stream>>>(ff1w, ff1_b, 2 * FFc * 64);
    cvt_bf16_kernel<<<(2 * 64 * FFc + 255) / 256, 256, 0, stream>>>(ff2w, ff2_b, 2 * 64 * FFc);
    addcvt_kernel<<<(Mc * Hc * INc + 255) / 256, 256, 0, stream>>>(W0, W1, w01_b, Mc * Hc * INc);
    bias3_kernel<<<1, 256, 0, stream>>>(b0, bl, b1, bsum, Mc * Hc);

    hipMemsetAsync(agg, 0, (size_t)Mc * Nn * INc * 4, stream);
    {
        long long total = (long long)Mc * Ec * 32;
        scatter_kernel<<<(unsigned)(total / 256), 256, 0, stream>>>(x, ei, agg);
    }
    gnn_kernel<<<(Mc * (Nn / 16)) / 4, 128, 0, stream>>>(x, agg, wl_b, w01_b, wo_b,
                                                         bsum, bo, mpw, ebuf);
    int tiles16 = (Nn * Mc) / 16;                 // 7500
    int tiles32 = (Nn * Mc) / 32;                 // 3750
    for (int l = 0; l < 2; ++l) {
        qkv_kernel<<<tiles16 / 4, 128, 0, stream>>>(ebuf, tinw_b, tinb, qkvb, l);
        attn_kernel<<<Nn / 4, 128, 0, stream>>>(qkvb, obuf);
        oproj_ln_kernel<<<tiles16 / 4, 128, 0, stream>>>(obuf, tow_b, tob, ln1g, ln1b, ebuf, l);
        ff_kernel<<<(tiles32 + 3) / 4, 128, 0, stream>>>(ff1_b, ff1b, ff2_b, ff2b,
                                                         ln2g, ln2b, ebuf, l);
    }
    head_kernel<<<(Nn + 255) / 256, 256, 0, stream>>>(ebuf, r1w, r1b, r2w, r2b, out);
}
